// VectorQuantizer_Instance_Vr_All_68685116998174
// MI455X (gfx1250) — matrix operand order repaired and checked by reading, not compiled, not run on hardware
//
#include <hip/hip_runtime.h>
#include <hip/hip_bf16.h>

#define B_ROWS 2048
#define E_ROWS 512
#define DIM    16384

typedef __attribute__((ext_vector_type(16))) __bf16 v16bf;
typedef __attribute__((ext_vector_type(8)))  float  v8f;

union Frag {
    uint4 u[2];
    v16bf v;
};

__device__ __forceinline__ unsigned fkey(float f) {
    unsigned u = __float_as_uint(f);
    return (u & 0x80000000u) ? ~u : (u | 0x80000000u);
}

__device__ __forceinline__ unsigned long long shfl_xor_u64(unsigned long long v, int mask) {
    unsigned lo = (unsigned)(v & 0xFFFFFFFFull);
    unsigned hi = (unsigned)(v >> 32);
    lo = __shfl_xor(lo, mask, 32);
    hi = __shfl_xor(hi, mask, 32);
    return ((unsigned long long)hi << 32) | (unsigned long long)lo;
}

// ---------------------------------------------------------------------------
// Kernel 0: initialize workspace state (deterministic, runs every launch)
// ---------------------------------------------------------------------------
__global__ void vq_init_kernel(unsigned long long* packed, unsigned* counts,
                               float* lossPartial) {
    int t = blockIdx.x * blockDim.x + threadIdx.x;
    if (t < B_ROWS) {
        packed[t] = 0xFFFFFFFFFFFFFFFFull;
        lossPartial[t] = 0.0f;
    }
    if (t < E_ROWS) counts[t] = 0u;
}

// ---------------------------------------------------------------------------
// Kernel 1: f32 -> bf16 (RNE) conversion + per-row squared norms.
// One block per row; rows [0,2048) = inputs, rows [2048,2560) = codebook.
// ---------------------------------------------------------------------------
__global__ __launch_bounds__(256) void vq_convert_norm_kernel(
    const float* __restrict__ inputs, const float* __restrict__ codebook,
    unsigned short* __restrict__ Abf, unsigned short* __restrict__ Ebf,
    float* __restrict__ xnorm, float* __restrict__ enorm) {
    int row = blockIdx.x;
    const float* src;
    unsigned short* dst;
    float* nrm;
    if (row < B_ROWS) {
        src = inputs + (size_t)row * DIM;
        dst = Abf + (size_t)row * DIM;
        nrm = &xnorm[row];
    } else {
        int r = row - B_ROWS;
        src = codebook + (size_t)r * DIM;
        dst = Ebf + (size_t)r * DIM;
        nrm = &enorm[r];
    }

    const float4* s4 = reinterpret_cast<const float4*>(src);
    ushort4* d4 = reinterpret_cast<ushort4*>(dst);
    float s = 0.0f;
#pragma unroll
    for (int j = 0; j < DIM / 4 / 256; ++j) {       // 16 iterations
        int i = threadIdx.x + j * 256;
        float4 v = s4[i];
        s += v.x * v.x + v.y * v.y + v.z * v.z + v.w * v.w;
        ushort4 o;
        unsigned u;
        u = __float_as_uint(v.x); o.x = (unsigned short)((u + 0x7FFFu + ((u >> 16) & 1u)) >> 16);
        u = __float_as_uint(v.y); o.y = (unsigned short)((u + 0x7FFFu + ((u >> 16) & 1u)) >> 16);
        u = __float_as_uint(v.z); o.z = (unsigned short)((u + 0x7FFFu + ((u >> 16) & 1u)) >> 16);
        u = __float_as_uint(v.w); o.w = (unsigned short)((u + 0x7FFFu + ((u >> 16) & 1u)) >> 16);
        d4[i] = o;
    }

    __shared__ float red[256];
    red[threadIdx.x] = s;
    __syncthreads();
    for (int off = 128; off > 0; off >>= 1) {
        if (threadIdx.x < off) red[threadIdx.x] += red[threadIdx.x + off];
        __syncthreads();
    }
    if (threadIdx.x == 0) *nrm = red[0];
}

// ---------------------------------------------------------------------------
// Kernel 2: bf16 WMMA GEMM (-2*x.e) with fused distance + row argmin.
// Block tile: 64 (M) x 128 (N); 8 wave32 waves arranged 2x4; each wave owns a
// 32x32 register tile = 4 accumulators -> 4x v_wmma_f32_16x16x32_bf16 / K-step.
// ---------------------------------------------------------------------------
__global__ __launch_bounds__(256, 1) void vq_gemm_argmin_kernel(
    const unsigned short* __restrict__ Abf, const unsigned short* __restrict__ Ebf,
    const float* __restrict__ xnorm, const float* __restrict__ enorm,
    unsigned long long* __restrict__ packedOut) {
    __shared__ float sXn[64];
    __shared__ float sEn[128];

    const int m_block = blockIdx.y * 64;
    const int n_block = blockIdx.x * 128;
    const int t = threadIdx.x;
    if (t < 64) sXn[t] = xnorm[m_block + t];
    if (t < 128) sEn[t] = enorm[n_block + t];
    __syncthreads();

    const int wave = t >> 5;
    const int lane = t & 31;
    const int waveM = wave >> 2;      // 0..1
    const int waveN = wave & 3;       // 0..3
    const int m0 = m_block + waveM * 32;
    const int n0 = n_block + waveN * 32;
    const int laneLo = lane & 15;
    const bool hiHalf = lane >= 16;

    // A fragment striping (16-bit A 16x32): lane<16 holds K{0..7,16..23},
    // lane>=16 holds K{8..15,24..31} for row M=lane%16.  uint4 == 8 bf16.
    const uint4* a0 = reinterpret_cast<const uint4*>(Abf + (size_t)(m0 + laneLo) * DIM);
    const uint4* a1 = reinterpret_cast<const uint4*>(Abf + (size_t)(m0 + 16 + laneLo) * DIM);
    // B fragment striping (16-bit B 32x16): lane n holds column n, K{0..7,16..23};
    // lane n+16 holds column n, K{8..15,24..31}.  Codebook rows are contiguous in K.
    const uint4* b0 = reinterpret_cast<const uint4*>(Ebf + (size_t)(n0 + laneLo) * DIM);
    const uint4* b1 = reinterpret_cast<const uint4*>(Ebf + (size_t)(n0 + 16 + laneLo) * DIM);

    const int aOff = hiHalf ? 1 : 0;  // +0/+2 (lo) vs +1/+3 (hi) uint4 units
    const int bOff = hiHalf ? 1 : 0;  // +0/+2 (lo) vs +1/+3 (hi)

    v8f acc00 = {}, acc01 = {}, acc10 = {}, acc11 = {};

    for (int k = 0; k < DIM; k += 32) {
        const int ku = k >> 3;        // uint4 index base
        Frag A0, A1, B0, B1;
        A0.u[0] = a0[ku + aOff]; A0.u[1] = a0[ku + aOff + 2];
        A1.u[0] = a1[ku + aOff]; A1.u[1] = a1[ku + aOff + 2];
        B0.u[0] = b0[ku + bOff]; B0.u[1] = b0[ku + bOff + 2];
        B1.u[0] = b1[ku + bOff]; B1.u[1] = b1[ku + bOff + 2];

        acc00 = __builtin_amdgcn_wmma_f32_16x16x32_bf16(false, A0.v, false, B0.v,
                                                        (short)0, acc00, false, false);
        acc01 = __builtin_amdgcn_wmma_f32_16x16x32_bf16(false, A0.v, false, B1.v,
                                                        (short)0, acc01, false, false);
        acc10 = __builtin_amdgcn_wmma_f32_16x16x32_bf16(false, A1.v, false, B0.v,
                                                        (short)0, acc10, false, false);
        acc11 = __builtin_amdgcn_wmma_f32_16x16x32_bf16(false, A1.v, false, B1.v,
                                                        (short)0, acc11, false, false);
    }

    // Epilogue: d = |x|^2 + |e|^2 - 2*dot ; pack (orderable_key, index) and
    // min-reduce across the 16 lanes holding one row, then global atomicMin.
    const int col0 = n0 + laneLo;          // ni = 0 column
    const int col1 = n0 + 16 + laneLo;     // ni = 1 column
    const float en0 = sEn[col0 - n_block];
    const float en1 = sEn[col1 - n_block];

#pragma unroll
    for (int mi = 0; mi < 2; ++mi) {
        v8f aA = mi ? acc10 : acc00;       // ni = 0
        v8f aB = mi ? acc11 : acc01;       // ni = 1
#pragma unroll
        for (int j = 0; j < 8; ++j) {
            const int rowLocal = waveM * 32 + mi * 16 + (hiHalf ? j + 8 : j);
            const float xn = sXn[rowLocal];
            const float d0 = xn + en0 - 2.0f * aA[j];
            const float d1 = xn + en1 - 2.0f * aB[j];
            unsigned long long p0 = ((unsigned long long)fkey(d0) << 32) | (unsigned)col0;
            unsigned long long p1 = ((unsigned long long)fkey(d1) << 32) | (unsigned)col1;
            unsigned long long p = p0 < p1 ? p0 : p1;
#pragma unroll
            for (int mask = 1; mask < 16; mask <<= 1) {
                unsigned long long q = shfl_xor_u64(p, mask);
                p = q < p ? q : p;
            }
            if (laneLo == 0)
                atomicMin(&packedOut[m_block + rowLocal], p);
        }
    }
}

// ---------------------------------------------------------------------------
// Kernel 3: gather codebook row (f32), straight-through output, per-row loss
// partial, one-hot encodings, integer histogram.  One block per input row.
// ---------------------------------------------------------------------------
__global__ __launch_bounds__(256) void vq_gather_kernel(
    const float* __restrict__ inputs, const float* __restrict__ codebook,
    const unsigned long long* __restrict__ packed,
    float* __restrict__ outQuant, float* __restrict__ outEnc,
    unsigned* __restrict__ counts, float* __restrict__ lossPartial) {
    const int b = blockIdx.x;
    const unsigned idx = (unsigned)(packed[b] & 0xFFFFFFFFull);

    const float4* cb = reinterpret_cast<const float4*>(codebook + (size_t)idx * DIM);
    const float4* in = reinterpret_cast<const float4*>(inputs + (size_t)b * DIM);
    float4* q = reinterpret_cast<float4*>(outQuant + (size_t)b * DIM);

    float s = 0.0f;
#pragma unroll
    for (int j = 0; j < DIM / 4 / 256; ++j) {       // 16 iterations
        int i = threadIdx.x + j * 256;
        float4 c = cb[i];
        float4 x = in[i];
        q[i] = c;
        float dx = c.x - x.x, dy = c.y - x.y, dz = c.z - x.z, dw = c.w - x.w;
        s += dx * dx + dy * dy + dz * dz + dw * dw;
    }

    for (int e = threadIdx.x; e < E_ROWS; e += 256)
        outEnc[(size_t)b * E_ROWS + e] = (e == (int)idx) ? 1.0f : 0.0f;

    if (threadIdx.x == 0) atomicAdd(&counts[idx], 1u);

    __shared__ float red[256];
    red[threadIdx.x] = s;
    __syncthreads();
    for (int off = 128; off > 0; off >>= 1) {
        if (threadIdx.x < off) red[threadIdx.x] += red[threadIdx.x + off];
        __syncthreads();
    }
    if (threadIdx.x == 0) lossPartial[b] = red[0];
}

// ---------------------------------------------------------------------------
// Kernel 4: finalize loss (deterministic sum of partials) and perplexity.
// ---------------------------------------------------------------------------
__global__ __launch_bounds__(512) void vq_finalize_kernel(
    const float* __restrict__ lossPartial, const unsigned* __restrict__ counts,
    float* __restrict__ outLoss, float* __restrict__ outPerp) {
    __shared__ float red[512];
    const int t = threadIdx.x;

    float s = 0.0f;
    for (int i = t; i < B_ROWS; i += 512) s += lossPartial[i];
    red[t] = s;
    __syncthreads();
    for (int off = 256; off > 0; off >>= 1) {
        if (t < off) red[t] += red[t + off];
        __syncthreads();
    }
    if (t == 0)
        *outLoss = red[0] * 1.25f / ((float)B_ROWS * (float)DIM);
    __syncthreads();

    float p = (float)counts[t] * (1.0f / (float)B_ROWS);
    red[t] = p * logf(p + 1e-10f);
    __syncthreads();
    for (int off = 256; off > 0; off >>= 1) {
        if (t < off) red[t] += red[t + off];
        __syncthreads();
    }
    if (t == 0) *outPerp = expf(-red[0]);
}

// ---------------------------------------------------------------------------
extern "C" void kernel_launch(void* const* d_in, const int* in_sizes, int n_in,
                              void* d_out, int out_size, void* d_ws, size_t ws_size,
                              hipStream_t stream) {
    const float* inputs   = (const float*)d_in[0];   // [2048, 32, 8, 8, 8] f32
    const float* codebook = (const float*)d_in[1];   // [512, 16384] f32

    // Output tuple flattened: loss | quantized_st (2048*16384) | perplexity |
    // encodings (2048*512)
    float* out      = (float*)d_out;
    float* outLoss  = out;
    float* outQuant = out + 1;
    float* outPerp  = out + 1 + (size_t)B_ROWS * DIM;
    float* outEnc   = outPerp + 1;

    // Workspace carve-up (~80.1 MB)
    char* ws = (char*)d_ws;
    unsigned short* Abf = (unsigned short*)ws;                       // 64 MB
    unsigned short* Ebf = Abf + (size_t)B_ROWS * DIM;                // 16 MB
    float* xnorm = (float*)(Ebf + (size_t)E_ROWS * DIM);             // 8 KB
    float* enorm = xnorm + B_ROWS;                                   // 2 KB
    unsigned long long* packed = (unsigned long long*)(enorm + E_ROWS); // 16 KB
    unsigned* counts = (unsigned*)(packed + B_ROWS);                 // 2 KB
    float* lossPartial = (float*)(counts + E_ROWS);                  // 8 KB

    vq_init_kernel<<<(B_ROWS + 255) / 256, 256, 0, stream>>>(packed, counts, lossPartial);

    vq_convert_norm_kernel<<<B_ROWS + E_ROWS, 256, 0, stream>>>(
        inputs, codebook, Abf, Ebf, xnorm, enorm);

    dim3 grid(E_ROWS / 128, B_ROWS / 64);   // (4, 32)
    vq_gemm_argmin_kernel<<<grid, 256, 0, stream>>>(Abf, Ebf, xnorm, enorm, packed);

    vq_gather_kernel<<<B_ROWS, 256, 0, stream>>>(
        inputs, codebook, packed, outQuant, outEnc, counts, lossPartial);

    vq_finalize_kernel<<<1, 512, 0, stream>>>(lossPartial, counts, outLoss, outPerp);
}
